// CausalSelfAttention_90512140796643
// MI455X (gfx1250) — matrix operand order repaired and checked by reading, not compiled, not run on hardware
//
#include <hip/hip_runtime.h>
#include <hip/hip_bf16.h>
#include <math.h>

// ---------------------------------------------------------------------------
// CDNA5 (gfx1250) causal self-attention, bf16 WMMA (16x16x32) + f32 accum.
// bf16 workspace: one-time f32->bf16 casts of x/W; Q/K/V/Y kept bf16 so the
// hot GEMM loops do pure vector loads with no per-iteration conversions.
// Wave32-only code: every WMMA block is exactly one wave (blockDim = 32).
// ---------------------------------------------------------------------------

typedef __attribute__((ext_vector_type(16))) __bf16 v16bf;
typedef __attribute__((ext_vector_type(8)))  __bf16 v8bf;
typedef __attribute__((ext_vector_type(8)))  float  v8f;

#define T_SEQ  2048
#define DMODEL 1024
#define NHEAD  16
#define HDIM   64
#define RMS_EPS 1.1920928955078125e-07f

static __device__ __forceinline__ v8f v8f_zero() {
  v8f z = {0.f, 0.f, 0.f, 0.f, 0.f, 0.f, 0.f, 0.f};
  return z;
}

static __device__ __forceinline__ v8f wmma_bf16(v16bf a, v16bf b, v8f c) {
  // D = A(16x32 bf16) * B(32x16 bf16) + C(16x16 f32)
  return __builtin_amdgcn_wmma_f32_16x16x32_bf16(
      /*neg_a=*/false, a, /*neg_b=*/false, b,
      /*c_mod=*/(short)0, c, /*reuse_a=*/false, /*reuse_b=*/false);
}

// A-matrix 16x32 bf16 fragment from row-major bf16 X (leading dim ld):
// lane l<16  : M=l,    K = {0..7, 16..23}
// lane l>=16 : M=l-16, K = {8..15, 24..31}
static __device__ __forceinline__ v16bf load_a_bf16(const __bf16* __restrict__ X,
                                                    int ld, int m0, int k0) {
  const int lane = threadIdx.x & 31;
  const int half = lane >> 4;
  const int lq   = lane & 15;
  const __bf16* base = X + (size_t)(m0 + lq) * ld + k0 + half * 8;
  v8bf lo = *(const v8bf*)(base);        // 16B aligned (k0 mult 32, half*8)
  v8bf hi = *(const v8bf*)(base + 16);
  return __builtin_shufflevector(lo, hi, 0, 1, 2, 3, 4, 5, 6, 7,
                                 8, 9, 10, 11, 12, 13, 14, 15);
}

// Same A-frag layout but sourced from an f32 array (used for the LDS P tile).
static __device__ __forceinline__ v16bf load_a_f32(const float* __restrict__ X,
                                                   int ld, int m0, int k0) {
  const int lane = threadIdx.x & 31;
  const int half = lane >> 4;
  const int lq   = lane & 15;
  const float* base = X + (size_t)(m0 + lq) * ld + k0 + half * 8;
  float f[16];
  *(float4*)(f + 0)  = *(const float4*)(base + 0);
  *(float4*)(f + 4)  = *(const float4*)(base + 4);
  *(float4*)(f + 8)  = *(const float4*)(base + 16);
  *(float4*)(f + 12) = *(const float4*)(base + 20);
  v16bf a;
#pragma unroll
  for (int i = 0; i < 16; ++i) a[i] = (__bf16)f[i];
  return a;
}

// B-matrix 32x16 bf16 fragment where B[kk][n] = R[n0+n][k0+kk]
// (rows of row-major R are columns of B; x@W^T, Q K^T, and P@V via Vt).
// lane layout: N = lane%16, K = {half*8..+7, 16+half*8..+7} -> two 16B loads.
static __device__ __forceinline__ v16bf load_b_rows(const __bf16* __restrict__ R,
                                                    int ld, int n0, int k0) {
  const int lane = threadIdx.x & 31;
  const int half = lane >> 4;
  const int lq   = lane & 15;
  const __bf16* base = R + (size_t)(n0 + lq) * ld + k0 + half * 8;
  v8bf lo = *(const v8bf*)(base);        // 16B aligned (k0 mult 32, half*8)
  v8bf hi = *(const v8bf*)(base + 16);
  return __builtin_shufflevector(lo, hi, 0, 1, 2, 3, 4, 5, 6, 7,
                                 8, 9, 10, 11, 12, 13, 14, 15);
}

// Reductions within a 16-lane half (C-layout rows live in one half).
static __device__ __forceinline__ float half_sum(float v) {
  v += __shfl_xor(v, 1, 32);
  v += __shfl_xor(v, 2, 32);
  v += __shfl_xor(v, 4, 32);
  v += __shfl_xor(v, 8, 32);
  return v;
}
static __device__ __forceinline__ float half_max(float v) {
  v = fmaxf(v, __shfl_xor(v, 1, 32));
  v = fmaxf(v, __shfl_xor(v, 2, 32));
  v = fmaxf(v, __shfl_xor(v, 4, 32));
  v = fmaxf(v, __shfl_xor(v, 8, 32));
  return v;
}

// ---------------------------------------------------------------------------
// One-time f32 -> bf16 cast (x and the four weight matrices). 8 elems/thread.
// ---------------------------------------------------------------------------
__global__ __launch_bounds__(256)
void cast_bf16_kernel(const float* __restrict__ src, __bf16* __restrict__ dst,
                      int n8) {
  const int i = blockIdx.x * blockDim.x + threadIdx.x;
  if (i >= n8) return;
  const float4* s = (const float4*)src + (size_t)i * 2;
  const float4 a = s[0];
  const float4 b = s[1];
  v8bf o;
  o[0] = (__bf16)a.x; o[1] = (__bf16)a.y; o[2] = (__bf16)a.z; o[3] = (__bf16)a.w;
  o[4] = (__bf16)b.x; o[5] = (__bf16)b.y; o[6] = (__bf16)b.z; o[7] = (__bf16)b.w;
  *((v8bf*)dst + i) = o;
}

// ---------------------------------------------------------------------------
// Kernel A: fused QKV projection + v-lerp + per-head RMS norm + RoPE.
// grid = (T/16, NHEAD, 3), block = 32. Each wave owns a 16(t) x 64(head) tile,
// i.e. a complete head row -> norm/rope fully in-register.
// Q/K written bf16 [H][T][64]; V written bf16 TRANSPOSED [H][64][T] so the
// attention PV B-fragment is two 16B loads along a Vt row.
// ---------------------------------------------------------------------------
__global__ __launch_bounds__(32)
void qkv_rope_kernel(const __bf16* __restrict__ xb, const float* __restrict__ vi,
                     const __bf16* __restrict__ Wqb, const __bf16* __restrict__ Wkb,
                     const __bf16* __restrict__ Wvb, const float* __restrict__ lamb,
                     __bf16* __restrict__ Q, __bf16* __restrict__ K,
                     __bf16* __restrict__ Vt) {
  const int t0    = blockIdx.x * 16;
  const int h     = blockIdx.y;
  const int which = blockIdx.z;
  const int n0    = h * HDIM;
  const __bf16* __restrict__ W = (which == 0) ? Wqb : (which == 1) ? Wkb : Wvb;

  const int lane = threadIdx.x & 31;
  const int half = lane >> 4;
  const int lq   = lane & 15;

  v8f acc[4];
#pragma unroll
  for (int a = 0; a < 4; ++a) acc[a] = v8f_zero();

  for (int kk = 0; kk < DMODEL; kk += 32) {
    if (kk + 32 < DMODEL) {   // CDNA5 global_prefetch_b8 for the next k-tile
      __builtin_prefetch(xb + (size_t)(t0 + lq) * DMODEL + kk + 32, 0, 3);
      __builtin_prefetch(W + (size_t)(n0 + lq) * DMODEL + kk + 32, 0, 3);
    }
    v16bf afrag = load_a_bf16(xb, DMODEL, t0, kk);
#pragma unroll
    for (int a = 0; a < 4; ++a) {
      v16bf bfrag = load_b_rows(W, DMODEL, n0 + a * 16, kk);
      acc[a] = wmma_bf16(afrag, bfrag, acc[a]);
    }
  }

  if (which == 2) {
    // V path: v = (1-lam)*v + lam*vi (f32), then bf16 store transposed.
    const float lam = lamb[0];
    const float oml = 1.0f - lam;
#pragma unroll
    for (int a = 0; a < 4; ++a)
#pragma unroll
      for (int j = 0; j < 8; ++j) {
        const int t = t0 + j + 8 * half;
        const int n = a * 16 + lq;
        const float v = oml * acc[a][j] + lam * vi[(size_t)t * DMODEL + n0 + n];
        Vt[((size_t)h * HDIM + n) * T_SEQ + t] = (__bf16)v;
      }
    return;
  }

  // RMS norm over the 64 head dims of each of the 16 rows.
  float rinv[8];
#pragma unroll
  for (int j = 0; j < 8; ++j) {
    float s = 0.f;
#pragma unroll
    for (int a = 0; a < 4; ++a) s += acc[a][j] * acc[a][j];
    s = half_sum(s);
    rinv[j] = rsqrtf(s * (1.0f / HDIM) + RMS_EPS);
  }
#pragma unroll
  for (int a = 0; a < 4; ++a)
#pragma unroll
    for (int j = 0; j < 8; ++j) acc[a][j] *= rinv[j];

  // RoPE half-split: dim n in [0,32) pairs with n+32  => acc[a] with acc[a+2].
#pragma unroll
  for (int a = 0; a < 2; ++a) {
    const float jrot = (float)(a * 16 + lq);
    const float invf = expf(jrot * (-9.210340371976184f / 32.0f)); // 10000^(-j/32)
#pragma unroll
    for (int j = 0; j < 8; ++j) {
      const float tpos = (float)(t0 + j + 8 * half);
      float sn, cs;
      sincosf(tpos * invf, &sn, &cs);
      const float x1 = acc[a][j];
      const float x2 = acc[a + 2][j];
      acc[a][j]     =  x1 * cs + x2 * sn;
      acc[a + 2][j] = -x1 * sn + x2 * cs;
    }
  }

  __bf16* __restrict__ OUT = (which == 0) ? Q : K;
#pragma unroll
  for (int a = 0; a < 4; ++a)
#pragma unroll
    for (int j = 0; j < 8; ++j) {
      const int t = t0 + j + 8 * half;
      OUT[((size_t)h * T_SEQ + t) * HDIM + a * 16 + lq] = (__bf16)acc[a][j];
    }
}

// ---------------------------------------------------------------------------
// Kernel B: flash-style causal attention with sink-sigmoid epilogue.
// grid = (T/16, NHEAD), block = 32. One wave per (head, 16 query rows).
// All operands bf16; softmax state f32. P transposed C-layout -> A-layout via
// a padded f32 LDS bounce (stride 36 floats kills the bank conflict, keeps
// 16B alignment). Y written bf16 [T][D] for the output projection.
// ---------------------------------------------------------------------------
__global__ __launch_bounds__(32)
void attn_kernel(const __bf16* __restrict__ Q, const __bf16* __restrict__ K,
                 const __bf16* __restrict__ Vt, const float* __restrict__ sinkw,
                 __bf16* __restrict__ Y) {
  const int t0 = blockIdx.x * 16;
  const int h  = blockIdx.y;
  const __bf16* __restrict__ Qh  = Q  + (size_t)h * T_SEQ * HDIM;
  const __bf16* __restrict__ Kh  = K  + (size_t)h * T_SEQ * HDIM;
  const __bf16* __restrict__ Vth = Vt + (size_t)h * HDIM * T_SEQ;

  const int lane = threadIdx.x & 31;
  const int half = lane >> 4;
  const int lq   = lane & 15;

  __shared__ __align__(16) float Pt[16 * 36];

  const v16bf aq0 = load_a_bf16(Qh, HDIM, t0, 0);
  const v16bf aq1 = load_a_bf16(Qh, HDIM, t0, 32);

  v8f yacc[4];
#pragma unroll
  for (int a = 0; a < 4; ++a) yacc[a] = v8f_zero();
  float mrow[8], lrow[8];
#pragma unroll
  for (int j = 0; j < 8; ++j) { mrow[j] = -1.0e30f; lrow[j] = 0.f; }

  const int kend = t0 + 16;            // exclusive bound on needed k indices
  for (int kc = 0; kc < kend; kc += 32) {
    // ---- S = Q K^T for 32 key columns (two 16x16 tiles, K-dim = hd = 64) --
    v8f s[2];
    s[0] = v8f_zero();
    s[1] = v8f_zero();
    {
      v16bf b;
      b = load_b_rows(Kh, HDIM, kc, 0);       s[0] = wmma_bf16(aq0, b, s[0]);
      b = load_b_rows(Kh, HDIM, kc, 32);      s[0] = wmma_bf16(aq1, b, s[0]);
      b = load_b_rows(Kh, HDIM, kc + 16, 0);  s[1] = wmma_bf16(aq0, b, s[1]);
      b = load_b_rows(Kh, HDIM, kc + 16, 32); s[1] = wmma_bf16(aq1, b, s[1]);
    }

    // ---- scale + causal mask --------------------------------------------
    const bool diag = (kc + 31 >= t0);
#pragma unroll
    for (int a = 0; a < 2; ++a)
#pragma unroll
      for (int j = 0; j < 8; ++j) {
        float sv = s[a][j] * 0.125f;          // 1/sqrt(64)
        if (diag) {
          const int kg = kc + a * 16 + lq;
          const int qg = t0 + j + 8 * half;
          if (kg > qg) sv = -1.0e30f;
        }
        s[a][j] = sv;
      }

    // ---- online softmax --------------------------------------------------
    float alpha[8];
#pragma unroll
    for (int j = 0; j < 8; ++j) {
      float rm = fmaxf(s[0][j], s[1][j]);
      rm = half_max(rm);
      const float mnew = fmaxf(mrow[j], rm);
      alpha[j] = expf(mrow[j] - mnew);
      mrow[j]  = mnew;
    }
#pragma unroll
    for (int a = 0; a < 2; ++a)
#pragma unroll
      for (int j = 0; j < 8; ++j)
        s[a][j] = expf(s[a][j] - mrow[j]);
#pragma unroll
    for (int j = 0; j < 8; ++j) {
      float rs = s[0][j] + s[1][j];
      rs = half_sum(rs);
      lrow[j] = lrow[j] * alpha[j] + rs;
    }
#pragma unroll
    for (int a = 0; a < 4; ++a)
#pragma unroll
      for (int j = 0; j < 8; ++j) yacc[a][j] *= alpha[j];

    // ---- transpose P (C-layout -> A-layout) through LDS ------------------
#pragma unroll
    for (int a = 0; a < 2; ++a)
#pragma unroll
      for (int j = 0; j < 8; ++j)
        Pt[(j + 8 * half) * 36 + a * 16 + lq] = s[a][j];
    __syncthreads();
    const v16bf pfrag = load_a_f32(Pt, 36, 0, 0);
    __syncthreads();

    // ---- Y += P @ V[kc:kc+32][:]  via Vt rows (two 16B loads per lane) -----
#pragma unroll
    for (int a = 0; a < 4; ++a) {
      v16bf bv = load_b_rows(Vth, T_SEQ, a * 16, kc);
      yacc[a] = wmma_bf16(pfrag, bv, yacc[a]);
    }
  }

  // ---- epilogue: lse, sink-sigmoid scale, normalize, store ---------------
  const float sw = sinkw[h];
  float fac[8];
#pragma unroll
  for (int j = 0; j < 8; ++j) {
    const float lse = mrow[j] + logf(lrow[j]);
    const float sig = 1.0f / (1.0f + expf(-(lse - sw)));
    fac[j] = sig / lrow[j];
  }
#pragma unroll
  for (int a = 0; a < 4; ++a)
#pragma unroll
    for (int j = 0; j < 8; ++j) {
      const int t = t0 + j + 8 * half;
      Y[(size_t)t * DMODEL + h * HDIM + a * 16 + lq] = (__bf16)(yacc[a][j] * fac[j]);
    }
}

// ---------------------------------------------------------------------------
// Kernel C: output projection  out = Y @ Wo^T (bf16 in, f32 out).
// grid = (T/16, D/64), block = 32.
// ---------------------------------------------------------------------------
__global__ __launch_bounds__(32)
void oproj_kernel(const __bf16* __restrict__ Y, const __bf16* __restrict__ Wob,
                  float* __restrict__ out) {
  const int t0 = blockIdx.x * 16;
  const int n0 = blockIdx.y * 64;

  const int lane = threadIdx.x & 31;
  const int half = lane >> 4;
  const int lq   = lane & 15;

  v8f acc[4];
#pragma unroll
  for (int a = 0; a < 4; ++a) acc[a] = v8f_zero();

  for (int kk = 0; kk < DMODEL; kk += 32) {
    if (kk + 32 < DMODEL) {
      __builtin_prefetch(Y   + (size_t)(t0 + lq) * DMODEL + kk + 32, 0, 3);
      __builtin_prefetch(Wob + (size_t)(n0 + lq) * DMODEL + kk + 32, 0, 3);
    }
    v16bf afrag = load_a_bf16(Y, DMODEL, t0, kk);
#pragma unroll
    for (int a = 0; a < 4; ++a) {
      v16bf bfrag = load_b_rows(Wob, DMODEL, n0 + a * 16, kk);
      acc[a] = wmma_bf16(afrag, bfrag, acc[a]);
    }
  }

#pragma unroll
  for (int a = 0; a < 4; ++a)
#pragma unroll
    for (int j = 0; j < 8; ++j) {
      const int t = t0 + j + 8 * half;
      out[(size_t)t * DMODEL + n0 + a * 16 + lq] = acc[a][j];
    }
}

// ---------------------------------------------------------------------------
extern "C" void kernel_launch(void* const* d_in, const int* in_sizes, int n_in,
                              void* d_out, int out_size, void* d_ws, size_t ws_size,
                              hipStream_t stream) {
  (void)in_sizes; (void)n_in; (void)out_size; (void)ws_size;
  const float* x     = (const float*)d_in[0];
  const float* vi    = (const float*)d_in[1];
  const float* Wq    = (const float*)d_in[2];
  const float* Wk    = (const float*)d_in[3];
  const float* Wv    = (const float*)d_in[4];
  const float* Wo    = (const float*)d_in[5];
  const float* lamb  = (const float*)d_in[6];
  const float* sinkw = (const float*)d_in[7];
  float* out = (float*)d_out;

  const size_t TD = (size_t)T_SEQ * DMODEL;   // 2M elements
  const size_t DD = (size_t)DMODEL * DMODEL;  // 1M elements

  __bf16* xb  = (__bf16*)d_ws;     // bf16 workspace, 28 MB total
  __bf16* Wqb = xb  + TD;
  __bf16* Wkb = Wqb + DD;
  __bf16* Wvb = Wkb + DD;
  __bf16* Wob = Wvb + DD;
  __bf16* Qb  = Wob + DD;          // [H][T][64]
  __bf16* Kb  = Qb  + TD;          // [H][T][64]
  __bf16* Vtb = Kb  + TD;          // [H][64][T] (transposed)
  __bf16* Yb  = Vtb + TD;          // [T][D]

  // One-time casts (8 elems per thread).
  const int c256 = 256;
  cast_bf16_kernel<<<(int)(TD / 8 + c256 - 1) / c256, c256, 0, stream>>>(x,  xb,  (int)(TD / 8));
  cast_bf16_kernel<<<(int)(DD / 8 + c256 - 1) / c256, c256, 0, stream>>>(Wq, Wqb, (int)(DD / 8));
  cast_bf16_kernel<<<(int)(DD / 8 + c256 - 1) / c256, c256, 0, stream>>>(Wk, Wkb, (int)(DD / 8));
  cast_bf16_kernel<<<(int)(DD / 8 + c256 - 1) / c256, c256, 0, stream>>>(Wv, Wvb, (int)(DD / 8));
  cast_bf16_kernel<<<(int)(DD / 8 + c256 - 1) / c256, c256, 0, stream>>>(Wo, Wob, (int)(DD / 8));

  dim3 gA(T_SEQ / 16, NHEAD, 3);
  qkv_rope_kernel<<<gA, 32, 0, stream>>>(xb, vi, Wqb, Wkb, Wvb, lamb, Qb, Kb, Vtb);

  dim3 gB(T_SEQ / 16, NHEAD);
  attn_kernel<<<gB, 32, 0, stream>>>(Qb, Kb, Vtb, sinkw, Yb);

  dim3 gC(T_SEQ / 16, DMODEL / 64);
  oproj_kernel<<<gC, 32, 0, stream>>>(Yb, Wob, out);
}
